// SimpleGNN_7421703488065
// MI455X (gfx1250) — matrix operand order repaired and checked by reading, not compiled, not run on hardware
//
#include <hip/hip_runtime.h>

#define NN 50000
#define NE 640000
#define FD 128
#define NTILES (NN / 16)  // 3125 exact

typedef __attribute__((ext_vector_type(16))) __bf16 v16bf;
typedef __attribute__((ext_vector_type(8)))  __bf16 v8bf;
typedef __attribute__((ext_vector_type(8)))  float  v8f;

// ---------------------------------------------------------------- zero ws
__global__ __launch_bounds__(256) void gnn_zero(float* __restrict__ p, int n) {
  int i = blockIdx.x * 256 + threadIdx.x;
  int s = gridDim.x * 256;
  for (; i < n; i += s) p[i] = 0.0f;
}

// ------------------------------------------------- edge scatter (segment sum)
// one wave32 per edge: each lane moves float4 (128 floats total per edge)
__global__ __launch_bounds__(256) void gnn_scatter(const float* __restrict__ x,
                                                   const long long* __restrict__ ei,
                                                   float* __restrict__ summed,
                                                   float* __restrict__ counts) {
  unsigned t = blockIdx.x * 256 + threadIdx.x;
  unsigned e = t >> 5;
  unsigned lane = threadIdx.x & 31;
  if (e >= NE) return;
  long long s = ei[e];
  long long d = ei[NE + e];
  float4 v = ((const float4*)(x + (size_t)s * FD))[lane];
  float* dp = summed + (size_t)d * FD + lane * 4;
  atomicAdd(dp + 0, v.x);
  atomicAdd(dp + 1, v.y);
  atomicAdd(dp + 2, v.z);
  atomicAdd(dp + 3, v.w);
  if (lane == 0) atomicAdd(counts + d, 1.0f);
}

// ------------------------------------------------------------- WMMA helpers
// A-fragment (16x32 bf16): lane m = lane&15, h = lane>>4.
// elems 0..15 <- row[kb + 16h + 0..15]
__device__ inline v16bf a_tile(const float* __restrict__ row, int kb, int h, float sc) {
  float4 q0 = *(const float4*)(row + kb + 16 * h);
  float4 q1 = *(const float4*)(row + kb + 16 * h + 4);
  float4 q2 = *(const float4*)(row + kb + 16 * h + 8);
  float4 q3 = *(const float4*)(row + kb + 16 * h + 12);
  v16bf a;
  a[0]  = (__bf16)(q0.x * sc); a[1]  = (__bf16)(q0.y * sc);
  a[2]  = (__bf16)(q0.z * sc); a[3]  = (__bf16)(q0.w * sc);
  a[4]  = (__bf16)(q1.x * sc); a[5]  = (__bf16)(q1.y * sc);
  a[6]  = (__bf16)(q1.z * sc); a[7]  = (__bf16)(q1.w * sc);
  a[8]  = (__bf16)(q2.x * sc); a[9]  = (__bf16)(q2.y * sc);
  a[10] = (__bf16)(q2.z * sc); a[11] = (__bf16)(q2.w * sc);
  a[12] = (__bf16)(q3.x * sc); a[13] = (__bf16)(q3.y * sc);
  a[14] = (__bf16)(q3.z * sc); a[15] = (__bf16)(q3.w * sc);
  return a;
}

// B-fragment (32x16 bf16): lane column n = lane&15; k = 16h + e, contiguous 16.
__device__ inline v16bf b_tile(const __bf16* __restrict__ bp) {
  v8bf lo = *(const v8bf*)bp;        // 16B aligned ds_load_b128
  v8bf hi = *(const v8bf*)(bp + 8);
  v16bf b;
#pragma unroll
  for (int i = 0; i < 8; ++i) { b[i] = lo[i]; b[8 + i] = hi[i]; }
  return b;
}

// ---------------------------------------------------- fused mean + 2x GEMM
// out = relu(mean_nb @ W_l + x @ W_r + b) ; one wave per 16-row tile
__global__ __launch_bounds__(256) void gnn_gemm(
    const float* __restrict__ summed, const float* __restrict__ counts,
    const float* __restrict__ x, const float* __restrict__ Wl,
    const float* __restrict__ Wr, const float* __restrict__ bias,
    float* __restrict__ out) {
  // transposed bf16 weights in LDS; +8 pad -> dword row stride 68 (bank stride 4)
  __shared__ __align__(16) __bf16 Wt[2][FD][FD + 8];
  // coalesced fill: each group of consecutive threads reads consecutive n.
  // one float4 load (fully coalesced b128) + 4 ds_store_b16 per group.
  for (int g = threadIdx.x; g < 2 * FD * (FD / 4); g += 256) {
    int mtx = g >> 12;            // 128*32 = 4096 groups per matrix
    int rem = g & 4095;
    int k = rem >> 5;             // 0..127
    int n4 = (rem & 31) * 4;      // 0,4,...,124
    const float* w = mtx ? Wr : Wl;
    float4 v = *(const float4*)(w + k * FD + n4);
    Wt[mtx][n4 + 0][k] = (__bf16)v.x;
    Wt[mtx][n4 + 1][k] = (__bf16)v.y;
    Wt[mtx][n4 + 2][k] = (__bf16)v.z;
    Wt[mtx][n4 + 3][k] = (__bf16)v.w;
  }
  __syncthreads();

  int lane = threadIdx.x & 31;
  int n16 = lane & 15;
  int h = lane >> 4;
  int wave = (blockIdx.x * 256 + threadIdx.x) >> 5;
  int nwaves = gridDim.x * 8;

  for (int tile = wave; tile < NTILES; tile += nwaves) {
    int row = tile * 16 + n16;  // A-row owned by this lane
    float inv = 1.0f / fmaxf(counts[row], 1.0f);
    const float* mrow = summed + (size_t)row * FD;
    const float* xrow = x + (size_t)row * FD;

    v16bf Am[4], Ax[4];
#pragma unroll
    for (int kt = 0; kt < 4; ++kt) {
      Am[kt] = a_tile(mrow, kt * 32, h, inv);   // mean = summed * 1/max(cnt,1)
      Ax[kt] = a_tile(xrow, kt * 32, h, 1.0f);  // self features
    }

    // keep nt loop rolled: caps VGPR pressure (~2 B-fragments live) so we get
    // 6+ waves/SIMD instead of 2, and avoids >256-VGPR s_set_vgpr_msb overhead
#pragma unroll 1
    for (int nt = 0; nt < 8; ++nt) {
      float bv = bias[nt * 16 + n16];
      v8f acc = {bv, bv, bv, bv, bv, bv, bv, bv};
#pragma unroll
      for (int kt = 0; kt < 4; ++kt) {
        v16bf bm = b_tile(&Wt[0][nt * 16 + n16][kt * 32 + 16 * h]);
        acc = __builtin_amdgcn_wmma_f32_16x16x32_bf16(false, Am[kt], false, bm,
                                                      (short)0, acc, false, false);
      }
#pragma unroll
      for (int kt = 0; kt < 4; ++kt) {
        v16bf bm = b_tile(&Wt[1][nt * 16 + n16][kt * 32 + 16 * h]);
        acc = __builtin_amdgcn_wmma_f32_16x16x32_bf16(false, Ax[kt], false, bm,
                                                      (short)0, acc, false, false);
      }
      // ReLU into distinct registers first, then store: lets the 8 b32 stores
      // issue as a clause instead of serializing on one data VGPR (s_wait_xcnt)
      float r[8];
#pragma unroll
      for (int i = 0; i < 8; ++i) r[i] = fmaxf(acc[i], 0.0f);
      float* orow = out + (size_t)(tile * 16) * FD + nt * 16 + n16;
#pragma unroll
      for (int i = 0; i < 8; ++i) {
        orow[(size_t)(i + 8 * h) * FD] = r[i];
      }
    }
  }
}

// ------------------------------------------------------------------ launch
extern "C" void kernel_launch(void* const* d_in, const int* in_sizes, int n_in,
                              void* d_out, int out_size, void* d_ws, size_t ws_size,
                              hipStream_t stream) {
  const float*     x  = (const float*)d_in[0];
  const long long* ei = (const long long*)d_in[1];
  const float*     Wl = (const float*)d_in[2];
  const float*     Wr = (const float*)d_in[3];
  const float*     b  = (const float*)d_in[4];
  float* out = (float*)d_out;

  float* summed = (float*)d_ws;                  // [NN, FD] f32
  float* counts = summed + (size_t)NN * FD;      // [NN] f32

  gnn_zero<<<1024, 256, 0, stream>>>(summed, NN * FD + NN);
  gnn_scatter<<<(NE * 32) / 256, 256, 0, stream>>>(x, ei, summed, counts);
  gnn_gemm<<<391, 256, 0, stream>>>(summed, counts, x, Wl, Wr, b, out);
}
